// MaskGenerator_12601434046919
// MI455X (gfx1250) — matrix operand order repaired and checked by reading, not compiled, not run on hardware
//
#include <hip/hip_runtime.h>
#include <stdint.h>

typedef __attribute__((ext_vector_type(16))) _Float16 v16h;
typedef __attribute__((ext_vector_type(8)))  float    v8f;
typedef __attribute__((ext_vector_type(8)))  uint32_t v8u;
typedef __attribute__((ext_vector_type(4)))  uint32_t u4v;
typedef __attribute__((ext_vector_type(4)))  float    f4v;

#define NODE_DIM 256
#define HID      128
#define H2DIM    64

// ---------------------------------------------------------------------------
// Prepack W1 (512x128 f32) and W2 (128x64 f32) into f16 WMMA B-fragment order.
// B fragment layout (v_wmma_*_16x16x32_f16, wave32):
//   VGPR v, half h : K = ktile*32 + (v<4 ? 0 : 16) + (lane<16 ? 0 : 8) + 2(v&3) + h ; N = lane&15
// Stored so each lane's 8 dwords are contiguous:
//   frag1[((ntile*16 + ktile)*32 + lane)*8 + v]   (8 ntiles x 16 ktiles)
//   frag2[((ntile*4  + ktile)*32 + lane)*8 + v]   (4 ntiles x  4 ktiles)
// ---------------------------------------------------------------------------
__global__ void prepack_w_frags(const float* __restrict__ W1,
                                const float* __restrict__ W2,
                                uint32_t* __restrict__ frag1,
                                uint32_t* __restrict__ frag2) {
  int t = blockIdx.x * blockDim.x + threadIdx.x;
  if (t < 32768) {                       // W1 fragments: 8*16*32*8 dwords
    int v    = t & 7;
    int lane = (t >> 3) & 31;
    int kt   = (t >> 8) & 15;
    int nt   = t >> 12;
    int k0   = kt * 32 + ((v < 4) ? 0 : 16) + ((lane < 16) ? 0 : 8) + 2 * (v & 3);
    int n    = nt * 16 + (lane & 15);
    _Float16 h0 = (_Float16)W1[(k0    ) * HID + n];
    _Float16 h1 = (_Float16)W1[(k0 + 1) * HID + n];
    uint32_t lo = (uint32_t)__builtin_bit_cast(unsigned short, h0);
    uint32_t hi = (uint32_t)__builtin_bit_cast(unsigned short, h1);
    frag1[t] = lo | (hi << 16);
  } else if (t < 36864) {                // W2 fragments: 4*4*32*8 dwords
    int t2   = t - 32768;
    int v    = t2 & 7;
    int lane = (t2 >> 3) & 31;
    int kt   = (t2 >> 8) & 3;
    int nt   = (t2 >> 10) & 3;
    int k0   = kt * 32 + ((v < 4) ? 0 : 16) + ((lane < 16) ? 0 : 8) + 2 * (v & 3);
    int n    = nt * 16 + (lane & 15);
    _Float16 h0 = (_Float16)W2[(k0    ) * H2DIM + n];
    _Float16 h1 = (_Float16)W2[(k0 + 1) * H2DIM + n];
    uint32_t lo = (uint32_t)__builtin_bit_cast(unsigned short, h0);
    uint32_t hi = (uint32_t)__builtin_bit_cast(unsigned short, h1);
    frag2[t2] = lo | (hi << 16);
  }
}

// Gather one A fragment (16x32 f16) lane-slice from an fp32 node row.
__device__ __forceinline__ v16h gather_a_frag(const float* row, int kA) {
  f4v f0 = *(const f4v*)(row + kA);
  f4v f1 = *(const f4v*)(row + kA + 4);
  f4v f2 = *(const f4v*)(row + kA + 16);
  f4v f3 = *(const f4v*)(row + kA + 20);
  v16h a;
  #pragma unroll
  for (int i = 0; i < 4; ++i) {
    a[i]      = (_Float16)f0[i];
    a[4 + i]  = (_Float16)f1[i];
    a[8 + i]  = (_Float16)f2[i];
    a[12 + i] = (_Float16)f3[i];
  }
  return a;
}

// ---------------------------------------------------------------------------
// Main kernel: one wave32 per PAIR of 16-edge tiles (32 edges), 8 waves/block.
// Software-pipelined: B fragment double-buffered (load p+1 issued before the
// WMMAs of p), A gathers hoisted one full k-tile ahead.
// ---------------------------------------------------------------------------
__global__ __launch_bounds__(256)
void edge_mlp_wmma(const float* __restrict__ node_emb,
                   const int*   __restrict__ edge_index, long long E,
                   const uint32_t* __restrict__ frag1,
                   const uint32_t* __restrict__ frag2,
                   const float* __restrict__ b1,
                   const float* __restrict__ b2,
                   const float* __restrict__ W3,
                   const float* __restrict__ b3,
                   float* __restrict__ out, int numPairs) {
  __shared__ __align__(16) char smem[8 * 8192];   // 8KB/wave: h1 (2x4KB f16) -> h2 (2x4KB f32)
  const int lane = threadIdx.x & 31;
  const int wave = threadIdx.x >> 5;
  const int pair = blockIdx.x * 8 + wave;
  if (pair >= numPairs) return;

  _Float16* h1a = (_Float16*)(smem + wave * 8192);          // tile0: 16x128 f16
  _Float16* h1b = (_Float16*)(smem + wave * 8192 + 4096);   // tile1: 16x128 f16
  float*    h2a = (float*)   (smem + wave * 8192);          // tile0: 16x64 f32 (reuse)
  float*    h2b = (float*)   (smem + wave * 8192 + 4096);   // tile1: 16x64 f32 (reuse)

  const int m16   = lane & 15;
  const int sel   = (lane < 16) ? 0 : 8;   // A-frag K shift for upper half-wave
  const int mBase = (lane < 16) ? 0 : 8;   // C-tile row base per half-wave

  long long e0 = (long long)pair * 32 + m16;        // tile0 edge for this lane's row
  long long e1 = e0 + 16;                           // tile1 edge
  if (e0 >= E) e0 = E - 1;
  if (e1 >= E) e1 = E - 1;
  const float* srcRow0 = node_emb + (long long)edge_index[e0]     * NODE_DIM;
  const float* dstRow0 = node_emb + (long long)edge_index[E + e0] * NODE_DIM;
  const float* srcRow1 = node_emb + (long long)edge_index[e1]     * NODE_DIM;
  const float* dstRow1 = node_emb + (long long)edge_index[E + e1] * NODE_DIM;

  const uint32_t* fragL1 = frag1 + lane * 8;   // + (nt*16+kt)*256 dwords
  const uint32_t* fragL2 = frag2 + lane * 8;   // + (nt*4 +kt)*256 dwords

  // ---- Layer 1: 16 k-tiles x 8 n-tiles x 2 edge-tiles -> 256 WMMA ----
  v8f acc0[8], acc1[8];
  #pragma unroll
  for (int nt = 0; nt < 8; ++nt) {
    float bv = b1[nt * 16 + m16];
    #pragma unroll
    for (int i = 0; i < 8; ++i) { acc0[nt][i] = bv; acc1[nt][i] = bv; }
  }

  // pipeline prologue: A frags for kt=0, B frag for (nt=0, kt=0)
  v16h a0 = gather_a_frag(srcRow0, sel);
  v16h a1 = gather_a_frag(srcRow1, sel);
  v8u  bw = *(const v8u*)(fragL1);

  #pragma unroll
  for (int kt = 0; kt < 16; ++kt) {
    // gather A for next k-tile (wraps on last iter; redundant loads, unused)
    const int ktn = (kt + 1) & 15;
    const int kAn = ((ktn & 7) * 32) + sel;
    v16h a0n = gather_a_frag((ktn < 8) ? srcRow0 : dstRow0, kAn);
    v16h a1n = gather_a_frag((ktn < 8) ? srcRow1 : dstRow1, kAn);
    #pragma unroll
    for (int nt = 0; nt < 8; ++nt) {
      // issue next B fragment before consuming current one
      const int pn  = (kt * 8 + nt + 1) & 127;    // next (kt,nt) position
      const int ktn2 = pn >> 3, ntn = pn & 7;
      v8u bwn = *(const v8u*)(fragL1 + (ntn * 16 + ktn2) * 256);
      v16h b = __builtin_bit_cast(v16h, bw);
      acc0[nt] = __builtin_amdgcn_wmma_f32_16x16x32_f16(
          false, a0, false, b, (short)0, acc0[nt], false, false);
      acc1[nt] = __builtin_amdgcn_wmma_f32_16x16x32_f16(
          false, a1, false, b, (short)0, acc1[nt], false, false);
      bw = bwn;
    }
    a0 = a0n;
    a1 = a1n;
  }

  // relu + stage h1 in LDS as f16 (C layout: row = r + mBase, col = nt*16 + m16)
  #pragma unroll
  for (int nt = 0; nt < 8; ++nt) {
    #pragma unroll
    for (int r = 0; r < 8; ++r) {
      float v0 = acc0[nt][r]; v0 = v0 > 0.0f ? v0 : 0.0f;
      float v1 = acc1[nt][r]; v1 = v1 > 0.0f ? v1 : 0.0f;
      h1a[(mBase + r) * HID + nt * 16 + m16] = (_Float16)v0;
      h1b[(mBase + r) * HID + nt * 16 + m16] = (_Float16)v1;
    }
  }
  // per-wave DS ops are processed in order -> no barrier needed

  // ---- Layer 2: 4 k-tiles x 4 n-tiles x 2 edge-tiles -> 32 WMMA ----
  v8f acc2a[4], acc2b[4];
  #pragma unroll
  for (int nt = 0; nt < 4; ++nt) {
    float bv = b2[nt * 16 + m16];
    #pragma unroll
    for (int i = 0; i < 8; ++i) { acc2a[nt][i] = bv; acc2b[nt][i] = bv; }
  }

  v8u bw2 = *(const v8u*)(fragL2);   // (nt=0, kt=0)
  #pragma unroll
  for (int kt = 0; kt < 4; ++kt) {
    const _Float16* ar0 = h1a + m16 * HID + kt * 32 + sel;
    const _Float16* ar1 = h1b + m16 * HID + kt * 32 + sel;
    u4v lo0 = *(const u4v*)(ar0);
    u4v hi0 = *(const u4v*)(ar0 + 16);
    u4v lo1 = *(const u4v*)(ar1);
    u4v hi1 = *(const u4v*)(ar1 + 16);
    v8u t0, t1;
    #pragma unroll
    for (int i = 0; i < 4; ++i) {
      t0[i] = lo0[i]; t0[4 + i] = hi0[i];
      t1[i] = lo1[i]; t1[4 + i] = hi1[i];
    }
    v16h a0l = __builtin_bit_cast(v16h, t0);
    v16h a1l = __builtin_bit_cast(v16h, t1);
    #pragma unroll
    for (int nt = 0; nt < 4; ++nt) {
      const int pn  = (kt * 4 + nt + 1) & 15;
      const int ktn = pn >> 2, ntn = pn & 3;
      v8u bwn = *(const v8u*)(fragL2 + (ntn * 4 + ktn) * 256);
      v16h b = __builtin_bit_cast(v16h, bw2);
      acc2a[nt] = __builtin_amdgcn_wmma_f32_16x16x32_f16(
          false, a0l, false, b, (short)0, acc2a[nt], false, false);
      acc2b[nt] = __builtin_amdgcn_wmma_f32_16x16x32_f16(
          false, a1l, false, b, (short)0, acc2b[nt], false, false);
      bw2 = bwn;
    }
  }

  // relu + stage h2 in LDS as f32 (overwrites h1 regions; DS in-order per wave)
  #pragma unroll
  for (int nt = 0; nt < 4; ++nt) {
    #pragma unroll
    for (int r = 0; r < 8; ++r) {
      float v0 = acc2a[nt][r]; v0 = v0 > 0.0f ? v0 : 0.0f;
      float v1 = acc2b[nt][r]; v1 = v1 > 0.0f ? v1 : 0.0f;
      h2a[(mBase + r) * H2DIM + nt * 16 + m16] = v0;
      h2b[(mBase + r) * H2DIM + nt * 16 + m16] = v1;
    }
  }

  // ---- Layer 3: 64 -> 1 dot + sigmoid, one lane per edge (all 32 lanes) ----
  {
    const float* h2p = (lane < 16) ? h2a : h2b;   // tile0 rows | tile1 rows
    float s = b3[0];
    #pragma unroll
    for (int j = 0; j < H2DIM; ++j) s += h2p[m16 * H2DIM + j] * W3[j];
    long long eo = (long long)pair * 32 + lane;   // lane<16 -> tile0, lane>=16 -> tile1
    if (eo < E) out[eo] = 1.0f / (1.0f + __expf(-s));
  }
}

// ---------------------------------------------------------------------------
extern "C" void kernel_launch(void* const* d_in, const int* in_sizes, int n_in,
                              void* d_out, int out_size, void* d_ws, size_t ws_size,
                              hipStream_t stream) {
  const float* node_emb   = (const float*)d_in[0];
  const int*   edge_index = (const int*)  d_in[1];
  const float* W1         = (const float*)d_in[2];
  const float* b1         = (const float*)d_in[3];
  const float* W2         = (const float*)d_in[4];
  const float* b2         = (const float*)d_in[5];
  const float* W3         = (const float*)d_in[6];
  const float* b3         = (const float*)d_in[7];
  float* out = (float*)d_out;

  const long long E = (long long)in_sizes[1] / 2;

  uint32_t* frag1 = (uint32_t*)d_ws;          // 32768 dwords = 128 KB
  uint32_t* frag2 = frag1 + 32768;            //  4096 dwords =  16 KB

  // pack W1/W2 into f16 WMMA B-fragment order
  prepack_w_frags<<<(36864 + 255) / 256, 256, 0, stream>>>(W1, W2, frag1, frag2);

  const int numPairs = (int)((E + 31) / 32);
  const int blocks   = (numPairs + 7) / 8;    // 8 waves (tile-pairs) per block
  edge_mlp_wmma<<<blocks, 256, 0, stream>>>(node_emb, edge_index, E,
                                            frag1, frag2, b1, b2, W3, b3,
                                            out, numPairs);
}
